// ProjectorLConv1d_53171695125244
// MI455X (gfx1250) — matrix operand order repaired and checked by reading, not compiled, not run on hardware
//
#include <hip/hip_runtime.h>
#include <hip/hip_bf16.h>

// ---------------- WMMA types ----------------
typedef __attribute__((ext_vector_type(16))) __bf16 v16bf;
typedef __attribute__((ext_vector_type(4)))  __bf16 v4bf;
typedef __attribute__((ext_vector_type(8)))  float  v8f;
typedef __attribute__((ext_vector_type(4)))  int    v4i;

union FragBF {
    v16bf v;
    uint4 q[2];
};

#define TM 128
#define TN 128
#define TK 32

// -------- async Global->LDS staging (CDNA5), with compile-safe fallback --------
#if __has_builtin(__builtin_amdgcn_global_load_async_to_lds_b128)
#define ASYNC_COPY16(ldsp, globp)                                              \
    __builtin_amdgcn_global_load_async_to_lds_b128(                            \
        (__attribute__((address_space(1))) v4i*)(globp),                       \
        (__attribute__((address_space(3))) v4i*)(ldsp), 0, 0)
#define ASYNC_WAIT() ASYNC_WAIT_IMPL()
__device__ __forceinline__ void ASYNC_WAIT_IMPL() {
#if __has_builtin(__builtin_amdgcn_s_wait_asynccnt)
    __builtin_amdgcn_s_wait_asynccnt(0);
#else
    asm volatile("s_wait_asynccnt 0" ::: "memory");
#endif
}
#else
#define ASYNC_COPY16(ldsp, globp) (*(uint4*)(ldsp) = *(const uint4*)(globp))
#define ASYNC_WAIT() ((void)0)
#endif

// ---------------- fp32 -> bf16 hi/lo split (layout preserved) ----------------
__global__ __launch_bounds__(256)
void split_kernel(const float* __restrict__ src, __bf16* __restrict__ hi,
                  __bf16* __restrict__ lo, int n4)
{
    int i = blockIdx.x * 256 + threadIdx.x;
    if (i >= n4) return;
    float4 f = ((const float4*)src)[i];
    float fv[4] = {f.x, f.y, f.z, f.w};
    v4bf h, l;
#pragma unroll
    for (int e = 0; e < 4; ++e) {
        __bf16 hb = (__bf16)fv[e];
        h[e] = hb;
        l[e] = (__bf16)(fv[e] - (float)hb);
    }
    ((v4bf*)hi)[i] = h;
    ((v4bf*)lo)[i] = l;
}

// -------- fp32 [K][N] -> bf16 hi/lo transposed [N][K] (tiled via LDS) --------
__global__ __launch_bounds__(256)
void split_t_kernel(const float* __restrict__ src, __bf16* __restrict__ hiT,
                    __bf16* __restrict__ loT, int K, int N)
{
    __shared__ float tile[32][33];
    const int n0 = blockIdx.x * 32, k0 = blockIdx.y * 32;
    const int tc = threadIdx.x & 31, tr8 = threadIdx.x >> 5;
#pragma unroll
    for (int i = 0; i < 4; ++i) {
        int k = tr8 + i * 8;
        tile[k][tc] = src[(size_t)(k0 + k) * N + n0 + tc];
    }
    __syncthreads();
#pragma unroll
    for (int i = 0; i < 4; ++i) {
        int n = tr8 + i * 8;
        float v = tile[tc][n];                 // element (k = tc, n)
        __bf16 hb = (__bf16)v;
        hiT[(size_t)(n0 + n) * K + k0 + tc] = hb;
        loT[(size_t)(n0 + n) * K + k0 + tc] = (__bf16)(v - (float)hb);
    }
}

// D = A(MxK) * B(KxN) + bias ; mode 1: out = res + relu(D)
// A given as pre-split hi/lo bf16 [M][K]; B pre-split AND pre-transposed [N][K].
__global__ __launch_bounds__(256)
void gemm_bf16x3_kernel(const __bf16* __restrict__ Ahi, const __bf16* __restrict__ Alo,
                        const __bf16* __restrict__ BhiT, const __bf16* __restrict__ BloT,
                        const float* __restrict__ bias, const float* __restrict__ res,
                        float* __restrict__ Cout, int M, int N, int K, int relu_mode)
{
    __shared__ __attribute__((aligned(16))) __bf16 sAhi[2][TM * TK];
    __shared__ __attribute__((aligned(16))) __bf16 sAlo[2][TM * TK];
    __shared__ __attribute__((aligned(16))) __bf16 sBhi[2][TN * TK];
    __shared__ __attribute__((aligned(16))) __bf16 sBlo[2][TN * TK];

    const int tid  = threadIdx.x;
    const int lane = tid & 31;
    const int wave = tid >> 5;
    const int l16  = lane & 15;
    const int h16  = (lane >> 4) & 1;
    const int wm   = wave & 3;   // M offset = 32*wm
    const int wn   = wave >> 2;  // N offset = 64*wn

    const int bM = blockIdx.y * TM;
    const int bN = blockIdx.x * TN;

    v8f acc[2][4];
#pragma unroll
    for (int mi = 0; mi < 2; ++mi)
#pragma unroll
        for (int ni = 0; ni < 4; ++ni)
#pragma unroll
            for (int e = 0; e < 8; ++e) acc[mi][ni][e] = 0.0f;

    // stage one 128x32 hi/lo pair for A and one 128x32 pair for B^T: 8 x b128 per thread
    auto stage = [&](int buf, int kk0) {
#pragma unroll
        for (int r2 = 0; r2 < 2; ++r2) {
            int idx = tid + r2 * 256;      // uint4 index over 512 per array
            int row = idx >> 2;            // 4 uint4 per 32-wide row
            int kc  = (idx & 3) * 8;       // bf16 column
            size_t ga = (size_t)(bM + row) * K + kk0 + kc;
            size_t gb = (size_t)(bN + row) * K + kk0 + kc;
            int la = row * TK + kc;
            ASYNC_COPY16(&sAhi[buf][la], Ahi + ga);
            ASYNC_COPY16(&sAlo[buf][la], Alo + ga);
            ASYNC_COPY16(&sBhi[buf][la], BhiT + gb);
            ASYNC_COPY16(&sBlo[buf][la], BloT + gb);
        }
    };

    const int nsteps = K / TK;
    stage(0, 0);
    ASYNC_WAIT();
    __syncthreads();

    for (int step = 0; step < nsteps; ++step) {
        const int cur = step & 1;
        if (step + 1 < nsteps) stage((step + 1) & 1, (step + 1) * TK);

        const __bf16* cAhi = sAhi[cur];
        const __bf16* cAlo = sAlo[cur];
        const __bf16* cBhi = sBhi[cur];
        const __bf16* cBlo = sBlo[cur];

        // A fragments (16-bit A 16x32: lanes 0-15 K 0..7/16..23, lanes 16-31 K 8..15/24..31)
        FragBF ahi[2], alo[2];
#pragma unroll
        for (int mi = 0; mi < 2; ++mi) {
            int arow = wm * 32 + mi * 16 + l16;
            const uint4* ph = (const uint4*)&cAhi[arow * TK + h16 * 8];
            ahi[mi].q[0] = ph[0];
            ahi[mi].q[1] = ph[2];
            const uint4* pl = (const uint4*)&cAlo[arow * TK + h16 * 8];
            alo[mi].q[0] = pl[0];
            alo[mi].q[1] = pl[2];
        }

#pragma unroll
        for (int ni = 0; ni < 4; ++ni) {
            // B fragment (16-bit B 32x16: lanes 0-15 K 0..7/16..23, lanes 16-31 K 8..15/24..31, N = lane%16)
            int bn = wn * 64 + ni * 16 + l16;
            FragBF bhi, blo;
            const uint4* pb = (const uint4*)&cBhi[bn * TK + h16 * 8];
            bhi.q[0] = pb[0];
            bhi.q[1] = pb[2];
            const uint4* pl = (const uint4*)&cBlo[bn * TK + h16 * 8];
            blo.q[0] = pl[0];
            blo.q[1] = pl[2];
#pragma unroll
            for (int mi = 0; mi < 2; ++mi) {
                acc[mi][ni] = __builtin_amdgcn_wmma_f32_16x16x32_bf16(
                    false, ahi[mi].v, false, bhi.v, (short)0, acc[mi][ni], false, false);
                acc[mi][ni] = __builtin_amdgcn_wmma_f32_16x16x32_bf16(
                    false, ahi[mi].v, false, blo.v, (short)0, acc[mi][ni], false, false);
                acc[mi][ni] = __builtin_amdgcn_wmma_f32_16x16x32_bf16(
                    false, alo[mi].v, false, bhi.v, (short)0, acc[mi][ni], false, false);
            }
        }

        if (step + 1 < nsteps) ASYNC_WAIT();
        __syncthreads();
    }

    // epilogue (C/D layout: VGPR e -> M = e + 8*h16, N = l16)
#pragma unroll
    for (int mi = 0; mi < 2; ++mi)
#pragma unroll
        for (int ni = 0; ni < 4; ++ni)
#pragma unroll
            for (int e = 0; e < 8; ++e) {
                int row = bM + wm * 32 + mi * 16 + h16 * 8 + e;
                int col = bN + wn * 64 + ni * 16 + l16;
                float v = acc[mi][ni][e] + bias[col];
                if (relu_mode) v = res[(size_t)row * N + col] + fmaxf(v, 0.0f);
                Cout[(size_t)row * N + col] = v;
            }
}

// ---------------- LayerNorm: one block per row ----------------
__global__ __launch_bounds__(256)
void layernorm_kernel(const float* __restrict__ hbuf, const float* __restrict__ g,
                      const float* __restrict__ b, float* __restrict__ y, int D)
{
    __shared__ float red[16];
    const int row = blockIdx.x;
    const float* hr = hbuf + (size_t)row * D;
    float s = 0.0f, s2 = 0.0f;
    for (int i = threadIdx.x; i < D; i += 256) {
        float v = hr[i];
        s += v; s2 += v * v;
    }
#pragma unroll
    for (int o = 16; o > 0; o >>= 1) {
        s  += __shfl_down(s,  o, 32);
        s2 += __shfl_down(s2, o, 32);
    }
    if ((threadIdx.x & 31) == 0) {
        red[(threadIdx.x >> 5) * 2]     = s;
        red[(threadIdx.x >> 5) * 2 + 1] = s2;
    }
    __syncthreads();
    if (threadIdx.x == 0) {
        float a = 0.0f, c = 0.0f;
        for (int w = 0; w < 8; ++w) { a += red[w * 2]; c += red[w * 2 + 1]; }
        red[0] = a; red[1] = c;
    }
    __syncthreads();
    const float mu   = red[0] / (float)D;
    const float var  = red[1] / (float)D - mu * mu;
    const float rstd = rsqrtf(var + 1e-5f);
    for (int i = threadIdx.x; i < D; i += 256)
        y[(size_t)row * D + i] = (hr[i] - mu) * rstd * g[i] + b[i];
}

// ---------------- wgen (y @ W_wgen + b) + per-head softmax over 5 taps ----------------
__global__ __launch_bounds__(256)
void wgen_softmax_kernel(const float* __restrict__ y, const float* __restrict__ Wg,
                         const float* __restrict__ bg, float* __restrict__ w, int D)
{
    const int t = blockIdx.x;
    const float* yr = y + (size_t)t * D;
    float acc[40];
#pragma unroll
    for (int n = 0; n < 40; ++n) acc[n] = 0.0f;
    const int kbase = threadIdx.x * 16;   // D=4096 / 256 threads
#pragma unroll 4
    for (int kk = 0; kk < 16; ++kk) {
        float yv = yr[kbase + kk];
        const float* wr = Wg + (size_t)(kbase + kk) * 40;
#pragma unroll
        for (int n = 0; n < 40; ++n) acc[n] += yv * wr[n];
    }
#pragma unroll
    for (int o = 16; o > 0; o >>= 1)
#pragma unroll
        for (int n = 0; n < 40; ++n) acc[n] += __shfl_down(acc[n], o, 32);

    __shared__ float red[8][40];
    const int lane = threadIdx.x & 31, wv = threadIdx.x >> 5;
    if (lane == 0)
#pragma unroll
        for (int n = 0; n < 40; ++n) red[wv][n] = acc[n];
    __syncthreads();

    if (threadIdx.x < 8) {   // one thread per head
        const int hd = threadIdx.x;
        float v[5];
#pragma unroll
        for (int k = 0; k < 5; ++k) {
            float sum = bg[hd * 5 + k];
            for (int p = 0; p < 8; ++p) sum += red[p][hd * 5 + k];
            v[k] = sum;
        }
        float mx = v[0];
#pragma unroll
        for (int k = 1; k < 5; ++k) mx = fmaxf(mx, v[k]);
        float se = 0.0f;
#pragma unroll
        for (int k = 0; k < 5; ++k) { v[k] = __expf(v[k] - mx); se += v[k]; }
        const float inv = 1.0f / se;
#pragma unroll
        for (int k = 0; k < 5; ++k) w[((size_t)t * 8 + hd) * 5 + k] = v[k] * inv;
    }
}

// ---------------- causal dynamic depthwise conv ----------------
__global__ __launch_bounds__(256)
void dynconv_kernel(const float* __restrict__ y, const float* __restrict__ w,
                    float* __restrict__ conv, int Tf, int D)
{
    const size_t idx = (size_t)blockIdx.x * 256 + threadIdx.x;  // over B*Tf*D
    const int d  = (int)(idx % D);
    const size_t bt = idx / D;
    const int tf = (int)(bt % Tf);
    const int b  = (int)(bt / Tf);
    const int hd = d >> 9;   // R = D/HEADS = 512
    const float* wp = w + ((size_t)(b * Tf + tf) * 8 + hd) * 5;
    float s = 0.0f;
#pragma unroll
    for (int k = 0; k < 5; ++k) {
        int tt = tf - 4 + k;
        if (tt >= 0) s += wp[k] * y[((size_t)b * Tf + tt) * D + d];
    }
    conv[idx] = s;
}

extern "C" void kernel_launch(void* const* d_in, const int* in_sizes, int n_in,
                              void* d_out, int out_size, void* d_ws, size_t ws_size,
                              hipStream_t stream) {
    const float* x      = (const float*)d_in[0];
    const float* W_proj = (const float*)d_in[1];
    const float* b_proj = (const float*)d_in[2];
    const float* ln_g   = (const float*)d_in[3];
    const float* ln_b   = (const float*)d_in[4];
    const float* W_wgen = (const float*)d_in[5];
    const float* b_wgen = (const float*)d_in[6];
    const float* W_ff   = (const float*)d_in[7];
    const float* b_ff   = (const float*)d_in[8];
    float* out = (float*)d_out;

    const int Bsz = 8, Tf = 512, D = 4096, K = 4096;
    const int M = Bsz * Tf;   // 4096

    char* ws = (char*)d_ws;
    float*  hbuf = (float*)ws;                              // 64 MB
    float*  ybuf = (float*)(ws + ((size_t)64  << 20));      // 64 MB
    float*  cbuf = (float*)(ws + ((size_t)128 << 20));      // 64 MB
    float*  wbuf = (float*)(ws + ((size_t)192 << 20));      // 640 KB
    __bf16* Shi  = (__bf16*)(ws + ((size_t)193 << 20));     // 32 MB (A hi)
    __bf16* Slo  = (__bf16*)(ws + ((size_t)225 << 20));     // 32 MB (A lo)
    __bf16* Thi  = (__bf16*)(ws + ((size_t)257 << 20));     // 32 MB (B^T hi)
    __bf16* Tlo  = (__bf16*)(ws + ((size_t)289 << 20));     // 32 MB (B^T lo)

    dim3 blk(256);
    dim3 gGemm(D / TN, M / TM);           // 32 x 32
    dim3 gT(D / 32, K / 32);              // 128 x 128
    const int n4 = M * K / 4;
    dim3 gSplit((n4 + 255) / 256);

    // 1) h = xf @ W_proj + b_proj
    split_kernel<<<gSplit, blk, 0, stream>>>(x, Shi, Slo, n4);
    split_t_kernel<<<gT, blk, 0, stream>>>(W_proj, Thi, Tlo, K, D);
    gemm_bf16x3_kernel<<<gGemm, blk, 0, stream>>>(Shi, Slo, Thi, Tlo, b_proj, nullptr,
                                                  hbuf, M, D, K, 0);
    // 2) y = LN(h) * g + b
    layernorm_kernel<<<dim3(M), blk, 0, stream>>>(hbuf, ln_g, ln_b, ybuf, D);
    // 3) w = softmax((y @ W_wgen + b_wgen).reshape(H,K), axis=K)
    wgen_softmax_kernel<<<dim3(M), blk, 0, stream>>>(ybuf, W_wgen, b_wgen, wbuf, D);
    // 4) conv[t] = sum_k w[t,k] * y[t-4+k]
    const size_t total = (size_t)M * D;
    dynconv_kernel<<<dim3((unsigned)((total + 255) / 256)), blk, 0, stream>>>(ybuf, wbuf, cbuf, Tf, D);
    // 5) out = h + relu(conv @ W_ff + b_ff)
    split_kernel<<<gSplit, blk, 0, stream>>>(cbuf, Shi, Slo, n4);
    split_t_kernel<<<gT, blk, 0, stream>>>(W_ff, Thi, Tlo, K, D);
    gemm_bf16x3_kernel<<<gGemm, blk, 0, stream>>>(Shi, Slo, Thi, Tlo, b_ff, hbuf,
                                                  out, M, D, K, 1);
}
